// MultiPrototypeHyperbolicClassifier_22531398434821
// MI455X (gfx1250) — matrix operand order repaired and checked by reading, not compiled, not run on hardware
//
#include <hip/hip_runtime.h>
#include <math.h>

// ---------------------------------------------------------------------------
// MultiPrototypeHyperbolicClassifier for MI455X (gfx1250, wave32, WMMA)
//
// Dominant cost: X[8192x1024] @ Zunit^T[1024x8000] = 134 GFLOP, compute-bound
// (AI ~ 1340 FLOP/B; HBM floor ~4.3us at 23.3 TB/s). f16 WMMA 16x16x32 with
// f32 accumulate gives 8x the K-depth of the f32 WMMA path; inputs are tiny,
// well-scaled (x ~ N(0,0.02), z unit rows). Whole f16 working set (~33 MB)
// fits in the 192 MB L2, so the GEMM streams from L2 after first touch.
//
// Data movement uses the CDNA5 async global->LDS path (ASYNCcnt) when the
// toolchain exposes it: zero staging VGPRs, double-buffered LDS tiles.
// ---------------------------------------------------------------------------

typedef _Float16 v16h __attribute__((ext_vector_type(16)));
typedef _Float16 h8   __attribute__((ext_vector_type(8)));
typedef _Float16 h4   __attribute__((ext_vector_type(4)));
typedef float    v8f  __attribute__((ext_vector_type(8)));
typedef int      v4i  __attribute__((ext_vector_type(4)));

static constexpr int Bn  = 8192;    // batch
static constexpr int Dn  = 1024;    // feature dim
static constexpr int Cn  = 1000;    // classes
static constexpr int CKn = 8000;    // classes * prototypes
static constexpr float EPSf = 1e-15f;

#if __has_builtin(__builtin_amdgcn_global_load_async_to_lds_b128)
#define HAVE_ASYNC_LDS 1
#else
#define HAVE_ASYNC_LDS 0
#endif

__device__ __forceinline__ void wait_asynccnt0() {
#if __has_builtin(__builtin_amdgcn_s_wait_asynccnt)
    __builtin_amdgcn_s_wait_asynccnt(0);
#else
    asm volatile("s_wait_asynccnt 0x0" ::: "memory");
#endif
}

// ---------------------------------------------------------------------------
// Prep: per row of a [rows x 1024] fp32 matrix compute sum of squares, then
// mode 0: store f16 copy + cx2[row] = |x|^2   (for x)
// mode 1: store f16 row/max(||row||,eps)      (unit prototype directions)
// ---------------------------------------------------------------------------
__global__ __launch_bounds__(256)
void hyp_prep_rows(const float* __restrict__ src, _Float16* __restrict__ dst,
                   float* __restrict__ cx2, int mode) {
    const int row = blockIdx.x;
    const int tid = threadIdx.x;
    const float4 v = *(const float4*)(src + (size_t)row * Dn + tid * 4);

    float ss = v.x * v.x + v.y * v.y + v.z * v.z + v.w * v.w;
#pragma unroll
    for (int m = 16; m >= 1; m >>= 1) ss += __shfl_xor(ss, m);

    __shared__ float red[8];
    const int wave = tid >> 5, lane = tid & 31;
    if (lane == 0) red[wave] = ss;
    __syncthreads();
    if (tid == 0) {
        float t = 0.f;
#pragma unroll
        for (int i = 0; i < 8; ++i) t += red[i];
        red[0] = t;
    }
    __syncthreads();
    const float total = red[0];

    float scale = 1.0f;
    if (mode == 0) {
        if (tid == 0) cx2[row] = total;
    } else {
        scale = 1.0f / fmaxf(sqrtf(total), EPSf);
    }

    h4 o;
    o[0] = (_Float16)(v.x * scale);
    o[1] = (_Float16)(v.y * scale);
    o[2] = (_Float16)(v.z * scale);
    o[3] = (_Float16)(v.w * scale);
    *(h4*)(dst + (size_t)row * Dn + tid * 4) = o;
}

// ---------------------------------------------------------------------------
// Fused WMMA GEMM + hyperbolic epilogue + logsumexp over K=8.
// Block tile 128x128, 8 waves (4M x 2N), wave tile 32x64 = 2x4 acc frags.
// K in steps of 32, double-buffered LDS (stride 40 halves: 16B-aligned,
// bank-conflict-free ds_load_b128 fragment gathers).
// OOB Z rows (ck >= 8000) are CLAMPED, not branched: a WMMA output column n
// depends only on B column n, and clamped columns map to class >= 1000 which
// is never stored -- so no divergence and no zero-fill needed.
// ---------------------------------------------------------------------------
__global__ __launch_bounds__(256)
void hyp_wmma_gemm(const _Float16* __restrict__ xh, const _Float16* __restrict__ zh,
                   const float* __restrict__ cx2, const float* __restrict__ wg,
                   const float* __restrict__ gammap, float* __restrict__ out) {
    constexpr int LDT = 40;  // LDS row stride in halves (32 + 8 pad)
    __shared__ __align__(16) _Float16 As[2][128 * LDT];
    __shared__ __align__(16) _Float16 Bs[2][128 * LDT];

    const int tid   = threadIdx.x;
    const int lane  = tid & 31;
    const int wave  = tid >> 5;
    const int waveM = wave & 3;   // 0..3 -> 32-row strip
    const int waveN = wave >> 2;  // 0..1 -> 64-col strip
    const int mBase = blockIdx.y * 128;
    const int nBase = blockIdx.x * 128;

    // global->LDS mapping: each thread moves 2x16B for A and for B per step
    int rowL[2], cgL[2], zrL[2];
#pragma unroll
    for (int t = 0; t < 2; ++t) {
        const int idx = tid + t * 256;
        rowL[t] = idx >> 2;          // 0..127
        cgL[t]  = (idx & 3) * 8;     // half offset within the 32-wide K slab
        const int zr = nBase + rowL[t];
        zrL[t] = zr < CKn ? zr : CKn - 1;  // clamp: harmless, see note above
    }

    v8f acc[2][4];
#pragma unroll
    for (int mi = 0; mi < 2; ++mi)
#pragma unroll
        for (int ni = 0; ni < 4; ++ni) acc[mi][ni] = (v8f){};

#if HAVE_ASYNC_LDS
    auto issueTile = [&](int k0, int buf) {
#pragma unroll
        for (int t = 0; t < 2; ++t) {
            __builtin_amdgcn_global_load_async_to_lds_b128(
                (v4i*)(xh + (size_t)(mBase + rowL[t]) * Dn + k0 + cgL[t]),
                (v4i*)&As[buf][rowL[t] * LDT + cgL[t]], 0, 0);
            __builtin_amdgcn_global_load_async_to_lds_b128(
                (v4i*)(zh + (size_t)zrL[t] * Dn + k0 + cgL[t]),
                (v4i*)&Bs[buf][rowL[t] * LDT + cgL[t]], 0, 0);
        }
    };
    issueTile(0, 0);
    wait_asynccnt0();
    __syncthreads();
#else
    float4 ra[2], rb[2];
    auto loadTile = [&](int k0) {
#pragma unroll
        for (int t = 0; t < 2; ++t) {
            ra[t] = *(const float4*)(xh + (size_t)(mBase + rowL[t]) * Dn + k0 + cgL[t]);
            rb[t] = *(const float4*)(zh + (size_t)zrL[t] * Dn + k0 + cgL[t]);
        }
    };
    auto stash = [&](int buf) {
#pragma unroll
        for (int t = 0; t < 2; ++t) {
            *(float4*)&As[buf][rowL[t] * LDT + cgL[t]] = ra[t];
            *(float4*)&Bs[buf][rowL[t] * LDT + cgL[t]] = rb[t];
        }
    };
    loadTile(0);
    stash(0);
    __syncthreads();
#endif

    // Fragment gathers (CDNA5 wave32 16-bit WMMA VGPR layouts):
    //  A (16x32): lane holds row m=lane%16; K halves {g..g+7, g+16..g+23},
    //             g = (lane/16)*8  -> two ds_load_b128.
    //  B (32x16): lane holds col n=lane%16; same K order as A:
    //             {g..g+7, g+16..g+23}, g = (lane/16)*8.
    const int rsel = lane & 15;
    const int gA   = (lane >> 4) * 8;
    const int kbB  = (lane >> 4) * 8;

    constexpr int KT = Dn / 32;  // 32 K-steps
    for (int kt = 0; kt < KT; ++kt) {
        const int buf = kt & 1;
#if HAVE_ASYNC_LDS
        if (kt + 1 < KT) issueTile((kt + 1) * 32, buf ^ 1);
#else
        if (kt + 1 < KT) loadTile((kt + 1) * 32);
#endif

        v16h a[2], b[4];
#pragma unroll
        for (int mi = 0; mi < 2; ++mi) {
            const int r = waveM * 32 + mi * 16 + rsel;
            const h8 lo = *(const h8*)&As[buf][r * LDT + gA];
            const h8 hi = *(const h8*)&As[buf][r * LDT + gA + 16];
            a[mi] = __builtin_shufflevector(lo, hi, 0, 1, 2, 3, 4, 5, 6, 7,
                                            8, 9, 10, 11, 12, 13, 14, 15);
        }
#pragma unroll
        for (int ni = 0; ni < 4; ++ni) {
            const int r = waveN * 64 + ni * 16 + rsel;
            const h8 lo = *(const h8*)&Bs[buf][r * LDT + kbB];
            const h8 hi = *(const h8*)&Bs[buf][r * LDT + kbB + 16];
            b[ni] = __builtin_shufflevector(lo, hi, 0, 1, 2, 3, 4, 5, 6, 7,
                                            8, 9, 10, 11, 12, 13, 14, 15);
        }
#pragma unroll
        for (int mi = 0; mi < 2; ++mi)
#pragma unroll
            for (int ni = 0; ni < 4; ++ni)
                acc[mi][ni] = __builtin_amdgcn_wmma_f32_16x16x32_f16(
                    false, a[mi], false, b[ni], (short)0, acc[mi][ni],
                    false, false);

#if HAVE_ASYNC_LDS
        if (kt + 1 < KT) wait_asynccnt0();  // own tile-(kt+1) async done
#else
        if (kt + 1 < KT) stash(buf ^ 1);
#endif
        __syncthreads();
    }

    // ---------------- epilogue: hyperbolic transform + logsumexp(K=8) -----
    const float drcr = 2.0f * gammap[0];
    const float ch = coshf(drcr);
    const float sh = sinhf(drcr);

#pragma unroll
    for (int mi = 0; mi < 2; ++mi) {
        // C/D layout: lanes 0-15 -> rows = vgpr idx, lanes 16-31 -> vgpr+8
        const int rBase = mBase + waveM * 32 + mi * 16 + ((lane >> 4) << 3);
        float cxv[8];
#pragma unroll
        for (int r = 0; r < 8; ++r) cxv[r] = cx2[rBase + r];

#pragma unroll
        for (int ni = 0; ni < 4; ++ni) {
            const int ck  = nBase + waveN * 64 + ni * 16 + rsel;
            const int cls = ck >> 3;
            const float g2 = 2.0f * wg[cls < Cn ? cls : Cn - 1];
#pragma unroll
            for (int r = 0; r < 8; ++r) {
                const float inner = acc[mi][ni][r];
                const float num = 2.0f * inner * ch - (1.0f + cxv[r]) * sh;
                const float den = fmaxf(1.0f - cxv[r], EPSf);
                const float l   = g2 * asinhf(num / den);
                // logsumexp across the 8 lanes of one class (cols 8c..8c+7)
                float m = l;
#pragma unroll
                for (int msk = 1; msk <= 4; msk <<= 1)
                    m = fmaxf(m, __shfl_xor(m, msk));
                float e = expf(l - m);
#pragma unroll
                for (int msk = 1; msk <= 4; msk <<= 1)
                    e += __shfl_xor(e, msk);
                const float lse = m + logf(e);
                if ((rsel & 7) == 0 && cls < Cn)
                    out[(size_t)(rBase + r) * Cn + cls] = lse;
            }
        }
    }
}

// ---------------------------------------------------------------------------
extern "C" void kernel_launch(void* const* d_in, const int* in_sizes, int n_in,
                              void* d_out, int out_size, void* d_ws, size_t ws_size,
                              hipStream_t stream) {
    (void)in_sizes; (void)n_in; (void)out_size; (void)ws_size;
    const float* x     = (const float*)d_in[0];  // [8192,1024]
    const float* wv    = (const float*)d_in[1];  // [1000,8,1024]
    const float* wg    = (const float*)d_in[2];  // [1000]
    const float* gamma = (const float*)d_in[3];  // scalar
    float* out = (float*)d_out;                  // [8192,1000]

    // workspace: xh (16 MB f16) | zh (16 MB f16) | cx2 (32 KB f32)
    _Float16* xh  = (_Float16*)d_ws;
    _Float16* zh  = xh + (size_t)Bn * Dn;
    float*    cx2 = (float*)(zh + (size_t)CKn * Dn);

    hyp_prep_rows<<<Bn, 256, 0, stream>>>(x, xh, cx2, 0);
    hyp_prep_rows<<<CKn, 256, 0, stream>>>(wv, zh, nullptr, 1);

    dim3 grid((CKn + 127) / 128, Bn / 128);  // 63 x 64
    hyp_wmma_gemm<<<grid, 256, 0, stream>>>(xh, zh, cx2, wg, gamma, out);
}
